// UniversalDeepfakeDetector_1039382085632
// MI455X (gfx1250) — matrix operand order repaired and checked by reading, not compiled, not run on hardware
//
#include <hip/hip_runtime.h>
#include <hip/hip_bf16.h>
#include <math.h>

// ---------------------------------------------------------------------------
// Types for CDNA5 WMMA (wave32, v_wmma_f32_16x16x32_bf16)
// ---------------------------------------------------------------------------
typedef __attribute__((ext_vector_type(16))) __bf16 v16bf;
typedef __attribute__((ext_vector_type(8)))  float  v8f;
typedef __attribute__((ext_vector_type(4)))  unsigned int u32x4;
typedef __attribute__((ext_vector_type(8)))  int i32x8;
typedef __attribute__((ext_vector_type(4)))  int i32x4;
typedef unsigned short u16;
typedef unsigned int   u32;

union BFVec { v16bf v; u16 s[16]; u32 u[8]; u32x4 q[2]; };

__device__ __forceinline__ u16 f2bf(float f) {
  u32 u = __float_as_uint(f);
  u32 r = u + 0x7FFFu + ((u >> 16) & 1u);   // round-to-nearest-even
  return (u16)(r >> 16);
}
__device__ __forceinline__ float bf2f(u16 h) {
  return __uint_as_float(((u32)h) << 16);
}

// A operand (16-bit 16x32 MxK) from row-major [M x K]:
// element e -> K = k0 + hi*16 + e : one contiguous 32-byte run.
__device__ __forceinline__ void load_a(BFVec& a, const u16* __restrict__ A,
                                       int lda, int row, int k0, int hi) {
  const u16* p = A + row * lda + k0 + hi * 16;
  a.q[0] = *(const u32x4*)(p);
  a.q[1] = *(const u32x4*)(p + 8);
}

// B operand (16-bit 32x16 KxN) from K-contiguous storage BT[N][K]:
// element e -> K = k0 + hi*16 + e : one contiguous 32-byte run.
__device__ __forceinline__ void load_bt(BFVec& b, const u16* __restrict__ BT,
                                        int ldb, int col, int k0, int hi) {
  const u16* p = BT + col * ldb + k0 + hi * 16;
  b.q[0] = *(const u32x4*)(p);
  b.q[1] = *(const u32x4*)(p + 8);
}

#define WMMA_BF16(a, b, c) \
  __builtin_amdgcn_wmma_f32_16x16x32_bf16(false, (a), false, (b), (short)0, (c), false, false)

// ---------------------------------------------------------------------------
// Tensor Data Mover: 2D bf16 tile Global -> LDS (D# built per ISA 8.3/8.4)
// ---------------------------------------------------------------------------
__device__ __forceinline__ void tdm_load_2d_bf16(unsigned lds_off, const void* gptr,
                                                 unsigned dim0, unsigned dim1,
                                                 unsigned tile0, unsigned tile1,
                                                 unsigned stride0) {
#if defined(__HIP_DEVICE_COMPILE__)
  unsigned long long ga = (unsigned long long)(uintptr_t)gptr;
  u32x4 g0;
  g0[0] = 1u;                                         // count=1, user mode
  g0[1] = lds_off;                                    // lds_addr (bytes)
  g0[2] = (u32)(ga & 0xffffffffu);                    // global_addr[31:0]
  g0[3] = (u32)((ga >> 32) & 0x01ffffffu) | 0x80000000u; // addr[56:32] | type=2
  i32x8 g1 = {};
  g1[0] = (int)(1u << 16);                            // data_size=1 -> 2 bytes
  g1[1] = (int)(dim0 << 16);                          // tensor_dim0[15:0]
  g1[2] = (int)((dim0 >> 16) | (dim1 << 16));         // dim0[31:16] | dim1[15:0]
  g1[3] = (int)((dim1 >> 16) | (tile0 << 16));        // dim1[31:16] | tile_dim0
  g1[4] = (int)(tile1 & 0xffffu);                     // tile_dim1 (tile_dim2=0)
  g1[5] = (int)stride0;                               // tensor_dim0_stride[31:0]
  i32x4 g2 = {};
  i32x4 g3 = {};
#if __clang_major__ >= 23
  i32x8 g4 = {};
  __builtin_amdgcn_tensor_load_to_lds(g0, g1, g2, g3, g4, 0);
#else
  __builtin_amdgcn_tensor_load_to_lds(g0, g1, g2, g3, 0);
#endif
  __builtin_amdgcn_s_wait_tensorcnt((short)0);
#else
  (void)lds_off; (void)gptr; (void)dim0; (void)dim1;
  (void)tile0; (void)tile1; (void)stride0;
#endif
}

// ---------------------------------------------------------------------------
// Constants
// ---------------------------------------------------------------------------
#define HW 384
#define PIX (HW * HW)        // 147456
#define NIMG 96              // 32 samples * 3 channels
#define NSAMP 32
#define C1OUT 64
#define C1PIX (192 * 192)    // 36864
#define POOLPIX (96 * 96)    // 9216
#define C2OUT 128
#define K1PAD 160            // 147 -> padded to mult of 32
#define K2 576               // 64*9

// ---------------------------------------------------------------------------
// Init / pack kernels
// ---------------------------------------------------------------------------
__global__ void init_dft(u16* __restrict__ dft) {   // [2][384*384] (Fr, Fi)
  int idx = blockIdx.x * 256 + threadIdx.x;
  if (idx >= PIX) return;
  int k = idx / HW, n = idx % HW;
  int m = (k * n) % HW;                             // exact phase index
  float a = (2.0f * 3.14159265358979323846f / (float)HW) * (float)m;
  dft[idx]       = f2bf(cosf(a));                   // Re(e^{-i a})
  dft[PIX + idx] = f2bf(-sinf(a));                  // Im(e^{-i a})
}

// Store x transposed per image: xbfT[img][w*384 + h] so the GEMM B operand
// (K = h contiguous) vectorizes to b128 loads.
__global__ void cast_x_T(const float* __restrict__ x, u16* __restrict__ xbfT) {
  int idx = blockIdx.x * 256 + threadIdx.x;
  if (idx >= NIMG * PIX) return;
  int img = idx / PIX, r = idx % PIX;
  int h = r / HW, w = r % HW;
  xbfT[(size_t)img * PIX + w * HW + h] = f2bf(x[idx]);
}

__global__ void pack_w1(const float* __restrict__ w, u16* __restrict__ w1p) {
  int idx = blockIdx.x * 256 + threadIdx.x;         // [64][160]
  if (idx >= C1OUT * K1PAD) return;
  int o = idx / K1PAD, kk = idx % K1PAD;
  w1p[idx] = (kk < 147) ? f2bf(w[o * 147 + kk]) : (u16)0;   // OIHW flat matches
}

__global__ void pack_w2(const float* __restrict__ w, u16* __restrict__ w2p) {
  int idx = blockIdx.x * 256 + threadIdx.x;         // [128][576], exact flat
  if (idx < C2OUT * K2) w2p[idx] = f2bf(w[idx]);
}

// ---------------------------------------------------------------------------
// DFT stage 1: per image, T = Fr * X, U = Fi * X (384^3 GEMMs, bf16 WMMA)
// blockIdx.y in [0,192): img = y>>1, which = y&1. grid.x = 576 tiles / 8 waves.
// ---------------------------------------------------------------------------
__global__ void dft_stage1(const u16* __restrict__ dft, const u16* __restrict__ xbfT,
                           u16* __restrict__ TU) {
  int batch = blockIdx.y;
  int img = batch >> 1, which = batch & 1;
  const u16* __restrict__ A  = dft + which * PIX;   // row-major [k][h]
  const u16* __restrict__ BT = xbfT + (size_t)img * PIX; // [w][h] = K-contig
  u16* __restrict__ Cm = TU + (size_t)batch * PIX;  // [img][which][k][w]
  int wave = threadIdx.x >> 5, lane = threadIdx.x & 31;
  int tile = blockIdx.x * 8 + wave;                 // 0..575
  int mi = tile / 24, ni = tile % 24;
  int row0 = mi * 16, col0 = ni * 16;
  int r = lane & 15, hi = lane >> 4;
  v8f acc = {};
  for (int k0 = 0; k0 < HW; k0 += 32) {
    __builtin_prefetch(A + (row0 + r) * HW + k0 + 32, 0, 1);
    BFVec a, b;
    load_a(a, A, HW, row0 + r, k0, hi);
    load_bt(b, BT, HW, col0 + r, k0, hi);
    acc = WMMA_BF16(a.v, b.v, acc);
  }
  int mb = hi * 8;
#pragma unroll
  for (int i = 0; i < 8; ++i)
    Cm[(row0 + mb + i) * HW + col0 + r] = f2bf(acc[i]);
}

// ---------------------------------------------------------------------------
// DFT stage 2: Yr = T*Fr^T - U*Fi^T ; Yi = T*Fi^T + U*Fr^T (Fr,Fi symmetric,
// so row-major Fr/Fi already serve as K-contiguous B^T). Then fftshifted
// log-magnitude.
// ---------------------------------------------------------------------------
__global__ void dft_stage2(const u16* __restrict__ dft, const u16* __restrict__ TU,
                           float* __restrict__ mag) {
  int img = blockIdx.y;
  const u16* __restrict__ T  = TU + ((size_t)img * 2 + 0) * PIX;
  const u16* __restrict__ U  = TU + ((size_t)img * 2 + 1) * PIX;
  const u16* __restrict__ Fr = dft;
  const u16* __restrict__ Fi = dft + PIX;
  int wave = threadIdx.x >> 5, lane = threadIdx.x & 31;
  int tile = blockIdx.x * 8 + wave;
  int mi = tile / 24, ni = tile % 24;
  int row0 = mi * 16, col0 = ni * 16;
  int r = lane & 15, hi = lane >> 4;
  v8f a1 = {}, a2 = {}, ai = {};
  for (int k0 = 0; k0 < HW; k0 += 32) {
    __builtin_prefetch(T + (row0 + r) * HW + k0 + 32, 0, 1);
    __builtin_prefetch(U + (row0 + r) * HW + k0 + 32, 0, 1);
    BFVec aT, aU, bR, bI;
    load_a(aT, T, HW, row0 + r, k0, hi);
    load_a(aU, U, HW, row0 + r, k0, hi);
    load_bt(bR, Fr, HW, col0 + r, k0, hi);
    load_bt(bI, Fi, HW, col0 + r, k0, hi);
    a1 = WMMA_BF16(aT.v, bR.v, a1);
    a2 = WMMA_BF16(aU.v, bI.v, a2);
    ai = WMMA_BF16(aT.v, bI.v, ai);
    ai = WMMA_BF16(aU.v, bR.v, ai);
  }
  int mb = hi * 8;
#pragma unroll
  for (int i = 0; i < 8; ++i) {
    float yr = a1[i] - a2[i], yi = ai[i];
    float m = logf(sqrtf(yr * yr + yi * yi) + 1.0f);
    int gr = row0 + mb + i, gc = col0 + r;
    int sr = (gr + 192) % HW, sc = (gc + 192) % HW;   // fftshift (N even)
    mag[(size_t)img * PIX + sr * HW + sc] = m;
  }
}

// ---------------------------------------------------------------------------
// Per-image-channel mean/std, then normalize -> bf16
// ---------------------------------------------------------------------------
__global__ void mag_stats(const float* __restrict__ mag, float* __restrict__ stats) {
  int img = blockIdx.x;
  const float* p = mag + (size_t)img * PIX;
  float s = 0.f, s2 = 0.f;
  for (int i = threadIdx.x; i < PIX; i += 256) { float v = p[i]; s += v; s2 += v * v; }
  __shared__ float sh[256], sh2[256];
  sh[threadIdx.x] = s; sh2[threadIdx.x] = s2; __syncthreads();
  for (int o = 128; o > 0; o >>= 1) {
    if (threadIdx.x < o) { sh[threadIdx.x] += sh[threadIdx.x + o]; sh2[threadIdx.x] += sh2[threadIdx.x + o]; }
    __syncthreads();
  }
  if (threadIdx.x == 0) {
    float mean = sh[0] / (float)PIX;
    float var = fmaxf(sh2[0] / (float)PIX - mean * mean, 0.0f);
    stats[img * 2 + 0] = mean;
    stats[img * 2 + 1] = 1.0f / (sqrtf(var) + 1e-8f);
  }
}

__global__ void normalize_mag(const float* __restrict__ mag, const float* __restrict__ stats,
                              u16* __restrict__ freqbf) {
  int idx = blockIdx.x * 256 + threadIdx.x;
  if (idx >= NIMG * PIX) return;
  int img = idx / PIX;
  freqbf[idx] = f2bf((mag[idx] - stats[img * 2]) * stats[img * 2 + 1]);
}

// ---------------------------------------------------------------------------
// Conv1: implicit GEMM, M=64, N=36864, K=147(pad160), 7x7 s2 p3 + BN + ReLU.
// Weights (64x160 bf16 = 20KB) staged into LDS by the Tensor Data Mover,
// consumed as the WMMA A operand via ds_load_b128.
// grid: x = 9216 tiles / 8 waves = 1152, y = sample
// ---------------------------------------------------------------------------
__global__ void conv1_wmma(const u16* __restrict__ freqbf, const u16* __restrict__ w1p,
                           const float* __restrict__ cb, const float* __restrict__ g,
                           const float* __restrict__ bb, u16* __restrict__ out) {
  __shared__ u16 w1s[C1OUT * K1PAD];                // 20480 bytes
  if (threadIdx.x == 0) {
    tdm_load_2d_bf16((unsigned)(uintptr_t)&w1s[0], w1p,
                     K1PAD, C1OUT, K1PAD, C1OUT, K1PAD);
  }
  __syncthreads();

  int s = blockIdx.y;
  int wave = threadIdx.x >> 5, lane = threadIdx.x & 31;
  int tile = blockIdx.x * 8 + wave;                 // 0..9215
  int mi = tile / 2304, ni = tile % 2304;
  int row0 = mi * 16, col0 = ni * 16;
  int r = lane & 15, hi = lane >> 4;
  const u16* __restrict__ in = freqbf + (size_t)s * 3 * PIX;
  int p = col0 + r, oh = p / 192, ow = p % 192;
  v8f acc = {};
  for (int k0 = 0; k0 < K1PAD; k0 += 32) {
    BFVec a, b;
    load_a(a, w1s, K1PAD, row0 + r, k0, hi);        // LDS b128 reads
#pragma unroll
    for (int e = 0; e < 16; ++e) {
      int K = k0 + hi * 16 + e;
      u16 v = 0;
      if (K < 147) {
        int c = K / 49, rem = K % 49, kh = rem / 7, kw = rem % 7;
        int ih = oh * 2 - 3 + kh, iw = ow * 2 - 3 + kw;
        if ((unsigned)ih < (unsigned)HW && (unsigned)iw < (unsigned)HW)
          v = in[c * PIX + ih * HW + iw];
      }
      b.s[e] = v;
    }
    acc = WMMA_BF16(a.v, b.v, acc);
  }
  int mb = hi * 8;
  int po = col0 + r;
  float bnmul = rsqrtf(1.0f + 1e-5f);
#pragma unroll
  for (int i = 0; i < 8; ++i) {
    int oc = row0 + mb + i;
    float v = (acc[i] + cb[oc]) * (g[oc] * bnmul) + bb[oc];
    out[((size_t)s * C1OUT + oc) * C1PIX + po] = f2bf(fmaxf(v, 0.0f));
  }
}

// ---------------------------------------------------------------------------
// Maxpool 3x3 stride 2 pad 1: (32,64,192,192) -> (32,64,96,96)
// ---------------------------------------------------------------------------
__global__ void maxpool_k(const u16* __restrict__ in, u16* __restrict__ out) {
  int idx = blockIdx.x * 256 + threadIdx.x;
  if (idx >= NSAMP * C1OUT * POOLPIX) return;
  int p = idx % POOLPIX, ch = idx / POOLPIX;
  int oh = p / 96, ow = p % 96;
  float m = -INFINITY;
#pragma unroll
  for (int kh = 0; kh < 3; ++kh)
#pragma unroll
    for (int kw = 0; kw < 3; ++kw) {
      int ih = oh * 2 - 1 + kh, iw = ow * 2 - 1 + kw;
      if ((unsigned)ih < 192u && (unsigned)iw < 192u)
        m = fmaxf(m, bf2f(in[(size_t)ch * C1PIX + ih * 192 + iw]));
    }
  out[idx] = f2bf(m);
}

// ---------------------------------------------------------------------------
// Conv2: implicit GEMM, M=128, N=9216, K=576, 3x3 s1 p1 + BN + ReLU,
// fused per-tile column reduction -> partial[sample][128][576] (deterministic).
// grid: x = 4608 tiles / 8 waves = 576, y = sample
// ---------------------------------------------------------------------------
__global__ void conv2_wmma(const u16* __restrict__ pool, const u16* __restrict__ w2p,
                           const float* __restrict__ cb, const float* __restrict__ g,
                           const float* __restrict__ bb, float* __restrict__ partial) {
  int s = blockIdx.y;
  int wave = threadIdx.x >> 5, lane = threadIdx.x & 31;
  int tile = blockIdx.x * 8 + wave;                 // 0..4607
  int mi = tile / 576, ni = tile % 576;
  int row0 = mi * 16, col0 = ni * 16;
  int r = lane & 15, hi = lane >> 4;
  const u16* __restrict__ in = pool + (size_t)s * C1OUT * POOLPIX;
  int p = col0 + r, oh = p / 96, ow = p % 96;
  v8f acc = {};
  for (int k0 = 0; k0 < K2; k0 += 32) {
    __builtin_prefetch(w2p + (row0 + r) * K2 + k0 + 32, 0, 1);
    BFVec a, b;
    load_a(a, w2p, K2, row0 + r, k0, hi);
#pragma unroll
    for (int e = 0; e < 16; ++e) {
      int K = k0 + hi * 16 + e;
      int c = K / 9, rem = K % 9, kh = rem / 3, kw = rem % 3;
      int ih = oh - 1 + kh, iw = ow - 1 + kw;
      b.s[e] = ((unsigned)ih < 96u && (unsigned)iw < 96u)
                   ? in[c * POOLPIX + ih * 96 + iw] : (u16)0;
    }
    acc = WMMA_BF16(a.v, b.v, acc);
  }
  int mb = hi * 8;
  float bnmul = rsqrtf(1.0f + 1e-5f);
#pragma unroll
  for (int i = 0; i < 8; ++i) {
    int oc = row0 + mb + i;
    float v = (acc[i] + cb[oc]) * (g[oc] * bnmul) + bb[oc];
    v = fmaxf(v, 0.0f);
    // reduce over the 16 columns held by the 16 lanes of this half-wave
    for (int off = 1; off < 16; off <<= 1) v += __shfl_xor(v, off, 32);
    if ((lane & 15) == 0)
      partial[((size_t)s * C2OUT + oc) * 576 + ni] = v;
  }
}

__global__ void reduce_partial(const float* __restrict__ partial, float* __restrict__ freqsum) {
  int idx = blockIdx.x * 256 + threadIdx.x;         // 0..4095 (32*128)
  if (idx >= NSAMP * C2OUT) return;
  float s = 0.f;
  for (int j = 0; j < 576; ++j) s += partial[(size_t)idx * 576 + j];
  freqsum[idx] = s;
}

// ---------------------------------------------------------------------------
// RGB GAP (mean over HxW per image-channel)
// ---------------------------------------------------------------------------
__global__ void rgb_gap(const float* __restrict__ x, float* __restrict__ gap) {
  int img = blockIdx.x;                             // 0..95 = s*3+c
  const float* p = x + (size_t)img * PIX;
  float s = 0.f;
  for (int i = threadIdx.x; i < PIX; i += 256) s += p[i];
  __shared__ float sh[256];
  sh[threadIdx.x] = s; __syncthreads();
  for (int o = 128; o > 0; o >>= 1) {
    if (threadIdx.x < o) sh[threadIdx.x] += sh[threadIdx.x + o];
    __syncthreads();
  }
  if (threadIdx.x == 0) gap[img] = sh[0] / (float)PIX;
}

// ---------------------------------------------------------------------------
// Final fusion MLP: z(32,130) -> 256 -> 128 -> 2 (tiny; plain f32 FMA)
// ---------------------------------------------------------------------------
__global__ void mlp_kernel(const float* __restrict__ gap, const float* __restrict__ freqsum,
                           const float* __restrict__ wrgb, const float* __restrict__ brgb,
                           const float* __restrict__ w1, const float* __restrict__ b1,
                           const float* __restrict__ w2, const float* __restrict__ b2,
                           const float* __restrict__ w3, const float* __restrict__ b3,
                           float* __restrict__ out) {
  __shared__ float z[32 * 130];                     // reused for h2 (4096 <= 4160)
  __shared__ float h1[32 * 256];
  int tid = threadIdx.x;
  for (int i = tid; i < 32 * 130; i += 256) {
    int s = i / 130, j = i % 130;
    float v;
    if (j < 2) {
      v = brgb[j];
      for (int c = 0; c < 3; ++c) v += gap[s * 3 + c] * wrgb[c * 2 + j];
    } else {
      v = freqsum[s * 128 + (j - 2)] / (float)POOLPIX;
    }
    z[i] = v;
  }
  __syncthreads();
  for (int i = tid; i < 32 * 256; i += 256) {
    int s = i / 256, j = i % 256;
    float v = b1[j];
    for (int k = 0; k < 130; ++k) v += z[s * 130 + k] * w1[k * 256 + j];
    h1[i] = fmaxf(v, 0.0f);
  }
  __syncthreads();
  float* h2 = z;                                    // safe: z fully consumed
  for (int i = tid; i < 32 * 128; i += 256) {
    int s = i / 128, j = i % 128;
    float v = b2[j];
    for (int k = 0; k < 256; ++k) v += h1[s * 256 + k] * w2[k * 128 + j];
    h2[i] = fmaxf(v, 0.0f);
  }
  __syncthreads();
  if (tid < 64) {
    int s = tid >> 1, j = tid & 1;
    float v = b3[j];
    for (int k = 0; k < 128; ++k) v += h2[s * 128 + k] * w3[k * 2 + j];
    out[s * 2 + j] = v;
  }
}

// ---------------------------------------------------------------------------
// Host launcher
// ---------------------------------------------------------------------------
extern "C" void kernel_launch(void* const* d_in, const int* in_sizes, int n_in,
                              void* d_out, int out_size, void* d_ws, size_t ws_size,
                              hipStream_t stream) {
  (void)in_sizes; (void)n_in; (void)out_size; (void)ws_size;
  const float* x       = (const float*)d_in[0];
  const float* w_rgb   = (const float*)d_in[1];
  const float* b_rgb   = (const float*)d_in[2];
  const float* conv1_w = (const float*)d_in[3];
  const float* conv1_b = (const float*)d_in[4];
  const float* bn1_g   = (const float*)d_in[5];
  const float* bn1_b   = (const float*)d_in[6];
  const float* conv2_w = (const float*)d_in[7];
  const float* conv2_b = (const float*)d_in[8];
  const float* bn2_g   = (const float*)d_in[9];
  const float* bn2_b   = (const float*)d_in[10];
  const float* fc1_w   = (const float*)d_in[11];
  const float* fc1_b   = (const float*)d_in[12];
  const float* fc2_w   = (const float*)d_in[13];
  const float* fc2_b   = (const float*)d_in[14];
  const float* fc3_w   = (const float*)d_in[15];
  const float* fc3_b   = (const float*)d_in[16];
  float* out = (float*)d_out;

  char* ws = (char*)d_ws;
  size_t off = 0;
  auto alloc = [&](size_t bytes) -> char* {
    char* p = ws + off;
    off += (bytes + 255) & ~(size_t)255;
    return p;
  };
  u16*   dft     = (u16*)  alloc((size_t)2 * PIX * 2);
  u16*   xbfT    = (u16*)  alloc((size_t)NIMG * PIX * 2);
  u16*   TU      = (u16*)  alloc((size_t)NIMG * 2 * PIX * 2);
  float* mag     = (float*)alloc((size_t)NIMG * PIX * 4);
  float* stats   = (float*)alloc((size_t)NIMG * 2 * 4);
  u16*   freqbf  = (u16*)  alloc((size_t)NIMG * PIX * 2);
  u16*   w1p     = (u16*)  alloc((size_t)C1OUT * K1PAD * 2);
  u16*   conv1o  = (u16*)  alloc((size_t)NSAMP * C1OUT * C1PIX * 2);
  u16*   pool    = (u16*)  alloc((size_t)NSAMP * C1OUT * POOLPIX * 2);
  u16*   w2p     = (u16*)  alloc((size_t)C2OUT * K2 * 2);
  float* partial = (float*)alloc((size_t)NSAMP * C2OUT * 576 * 4);
  float* gap     = (float*)alloc((size_t)NIMG * 4);
  float* freqsum = (float*)alloc((size_t)NSAMP * C2OUT * 4);

  init_dft<<<(PIX + 255) / 256, 256, 0, stream>>>(dft);
  cast_x_T<<<(NIMG * PIX + 255) / 256, 256, 0, stream>>>(x, xbfT);
  pack_w1<<<(C1OUT * K1PAD + 255) / 256, 256, 0, stream>>>(conv1_w, w1p);
  pack_w2<<<(C2OUT * K2 + 255) / 256, 256, 0, stream>>>(conv2_w, w2p);

  dft_stage1<<<dim3(72, 192), 256, 0, stream>>>(dft, xbfT, TU);
  dft_stage2<<<dim3(72, NIMG), 256, 0, stream>>>(dft, TU, mag);
  mag_stats<<<NIMG, 256, 0, stream>>>(mag, stats);
  normalize_mag<<<(NIMG * PIX + 255) / 256, 256, 0, stream>>>(mag, stats, freqbf);

  conv1_wmma<<<dim3(1152, NSAMP), 256, 0, stream>>>(freqbf, w1p, conv1_b, bn1_g, bn1_b, conv1o);
  maxpool_k<<<(NSAMP * C1OUT * POOLPIX + 255) / 256, 256, 0, stream>>>(conv1o, pool);
  conv2_wmma<<<dim3(576, NSAMP), 256, 0, stream>>>(pool, w2p, conv2_b, bn2_g, bn2_b, partial);
  reduce_partial<<<16, 256, 0, stream>>>(partial, freqsum);

  rgb_gap<<<NIMG, 256, 0, stream>>>(x, gap);
  mlp_kernel<<<1, 256, 0, stream>>>(gap, freqsum, w_rgb, b_rgb, fc1_w, fc1_b,
                                    fc2_w, fc2_b, fc3_w, fc3_b, out);
}
